// Graph_Constructor2_35124242546910
// MI455X (gfx1250) — matrix operand order repaired and checked by reading, not compiled, not run on hardware
//
#include <hip/hip_runtime.h>
#include <hip/hip_bf16.h>

typedef __bf16 bf16_t;
typedef __attribute__((ext_vector_type(16))) __bf16 v16bf;
typedef __attribute__((ext_vector_type(8)))  __bf16 v8bf;
typedef __attribute__((ext_vector_type(8)))  float  v8f;
typedef __attribute__((ext_vector_type(4)))  int    v4i;

// AS-qualified int4 for the gfx1250 async global->LDS builtin
typedef __attribute__((address_space(1))) v4i as1_v4i;   // global
typedef __attribute__((address_space(3))) v4i as3_v4i;   // LDS

#define N_ROWS 8192
#define C_DIM  256
#define TOPK   17        // K+1 with K=16

// LDS staging for the big GEMM: 128 A-rows + 128 B-rows, full K depth,
// pitch padded to 264 bf16 (528 B) to rotate LDS banks between rows.
#define LDS_PITCH 264
#define TILE_ROWS 128
#define GEMM_LDS_BYTES (2 * TILE_ROWS * LDS_PITCH * (int)sizeof(bf16_t))

// gfx1250 async global->LDS path (guarded: falls back to reg-staged copies)
#if defined(__has_builtin)
#  if __has_builtin(__builtin_amdgcn_global_load_async_to_lds_b128) && \
      __has_builtin(__builtin_amdgcn_s_wait_asynccnt)
#    define USE_ASYNC_LDS 1
#  endif
#endif
#ifndef USE_ASYNC_LDS
#  define USE_ASYNC_LDS 0
#endif

// ---------------- conversion kernels ----------------
__global__ void k_convert_x(const float* __restrict__ x, bf16_t* __restrict__ xb, int n) {
  int i = blockIdx.x * blockDim.x + threadIdx.x;
  if (i < n) xb[i] = (bf16_t)x[i];
}

// linT[r][c] = lin[c][r]  (so the B-operand of x@lin reads contiguous rows)
__global__ void k_convert_linT(const float* __restrict__ lin, bf16_t* __restrict__ linT) {
  int i = blockIdx.x * blockDim.x + threadIdx.x;   // over 256*256
  int r = i >> 8;
  int c = i & 255;
  linT[i] = (bf16_t)lin[c * C_DIM + r];
}

// ---------------- WMMA fragment loaders (pitch-parameterized) ----------------
// A-matrix 16x32 bf16 layout: lane L -> row M=L&15;
//   VGPR0-3: K = (L>>4)*8 + [0..7], VGPR4-7: K = 16 + (L>>4)*8 + [0..7]
__device__ __forceinline__ v16bf load_a_frag(const bf16_t* __restrict__ base,
                                             int row0, int k0, int lane, int pitch) {
  int r   = row0 + (lane & 15);
  int seg = lane >> 4;
  const bf16_t* p = base + r * pitch + k0 + seg * 8;
  v8bf lo = *(const v8bf*)p;
  v8bf hi = *(const v8bf*)(p + 16);
  v16bf o;
#pragma unroll
  for (int i = 0; i < 8; ++i) { o[i] = lo[i]; o[i + 8] = hi[i]; }
  return o;
}

// B-matrix 32x16 bf16 layout: lane L -> column N=L&15;
//   VGPR0-3: K = (L>>4)*8 + [0..7], VGPR4-7: K = 16 + (L>>4)*8 + [0..7] of that column.
// B[k][n] = src[row0+n][k] with src row-major -> contiguous row read.
__device__ __forceinline__ v16bf load_b_frag(const bf16_t* __restrict__ base,
                                             int row0, int k0, int lane, int pitch) {
  int r   = row0 + (lane & 15);
  int seg = lane >> 4;
  const bf16_t* p = base + r * pitch + k0 + seg * 8;
  v8bf lo = *(const v8bf*)p;
  v8bf hi = *(const v8bf*)(p + 16);
  v16bf o;
#pragma unroll
  for (int i = 0; i < 8; ++i) { o[i] = lo[i]; o[i + 8] = hi[i]; }
  return o;
}

// ---------------- stage 1: h = x @ lin (bf16 WMMA) ----------------
__global__ void __launch_bounds__(256) k_gemm_h(const bf16_t* __restrict__ xb,
                                                const bf16_t* __restrict__ linT,
                                                bf16_t* __restrict__ hb) {
  int lane  = threadIdx.x & 31;
  int wave  = (blockIdx.x * blockDim.x + threadIdx.x) >> 5;
  int m0    = (wave >> 4) * 16;   // 512 m-tiles
  int n0    = (wave & 15) * 16;   // 16 n-tiles
  v8f acc = {0.f, 0.f, 0.f, 0.f, 0.f, 0.f, 0.f, 0.f};
#pragma unroll
  for (int k0 = 0; k0 < C_DIM; k0 += 32) {
    v16bf a = load_a_frag(xb,   m0, k0, lane, C_DIM);
    v16bf b = load_b_frag(linT, n0, k0, lane, C_DIM);
    acc = __builtin_amdgcn_wmma_f32_16x16x32_bf16(false, a, false, b,
                                                  (short)0, acc, false, false);
  }
#pragma unroll
  for (int v = 0; v < 8; ++v) {
    int m = m0 + v + ((lane >> 4) << 3);
    int n = n0 + (lane & 15);
    hb[m * C_DIM + n] = (bf16_t)acc[v];
  }
}

// ---------------- LDS staging helper ----------------
// Copies 256 rows x 256 cols of bf16 from global (row-major, ld=C_DIM) into
// LDS with LDS_PITCH. 256 threads, 32 chunks (of 8 bf16) per row.
__device__ __forceinline__ void stage_tile(const bf16_t* __restrict__ gbase, int rowBase,
                                           bf16_t* lbase, int tid) {
#pragma unroll
  for (int it = 0; it < 32; ++it) {
    int c   = it * 256 + tid;       // chunk id 0..8191
    int row = c >> 5;               // 0..255
    int col = (c & 31) * 8;         // 0..248
    const bf16_t* g = gbase + (size_t)(rowBase + row) * C_DIM + col;
    bf16_t* l = lbase + row * LDS_PITCH + col;
#if USE_ASYNC_LDS
    __builtin_amdgcn_global_load_async_to_lds_b128(
        (as1_v4i*)(const void*)g, (as3_v4i*)l, 0, 0);
#else
    *(v8bf*)l = *(const v8bf*)g;
#endif
  }
}

// ---------------- stage 2: A = relu(tanh(h @ h^T)) ----------------
// block = 8 waves -> 128x128 tile; wave -> 64x32 (4x2 sub-tiles)
// Operands staged in LDS (async-to-LDS on gfx1250), WMMA fed from ds_load.
__global__ void __launch_bounds__(256) k_gemm_sim(const bf16_t* __restrict__ hb,
                                                  float* __restrict__ out) {
  extern __shared__ bf16_t smem[];
  bf16_t* smA = smem;                               // 128 x LDS_PITCH
  bf16_t* smB = smem + TILE_ROWS * LDS_PITCH;       // 128 x LDS_PITCH

  int tid  = threadIdx.x;
  int lane = tid & 31;
  int wave = tid >> 5;
  int wm   = wave >> 2;            // 0..1
  int wn   = wave & 3;             // 0..3

  // A rows = block's M range, B rows = block's N range (B = h^T trick)
  stage_tile(hb, blockIdx.y * 128, smA, tid);
  stage_tile(hb, blockIdx.x * 128, smB, tid);
#if USE_ASYNC_LDS
  __builtin_amdgcn_s_wait_asynccnt(0);
#endif
  __syncthreads();

  int mLoc = wm * 64;              // local row base in smA
  int nLoc = wn * 32;              // local row base in smB

  v8f acc[4][2];
#pragma unroll
  for (int i = 0; i < 4; ++i)
#pragma unroll
    for (int j = 0; j < 2; ++j)
      acc[i][j] = (v8f){0.f, 0.f, 0.f, 0.f, 0.f, 0.f, 0.f, 0.f};

#pragma unroll
  for (int k0 = 0; k0 < C_DIM; k0 += 32) {
    v16bf a[4], b[2];
#pragma unroll
    for (int i = 0; i < 4; ++i) a[i] = load_a_frag(smA, mLoc + 16 * i, k0, lane, LDS_PITCH);
#pragma unroll
    for (int j = 0; j < 2; ++j) b[j] = load_b_frag(smB, nLoc + 16 * j, k0, lane, LDS_PITCH);
#pragma unroll
    for (int i = 0; i < 4; ++i)
#pragma unroll
      for (int j = 0; j < 2; ++j)
        acc[i][j] = __builtin_amdgcn_wmma_f32_16x16x32_bf16(false, a[i], false, b[j],
                                                            (short)0, acc[i][j],
                                                            false, false);
  }

  int mBase = blockIdx.y * 128 + mLoc;
  int nBase = blockIdx.x * 128 + nLoc;
#pragma unroll
  for (int i = 0; i < 4; ++i)
#pragma unroll
    for (int j = 0; j < 2; ++j)
#pragma unroll
      for (int v = 0; v < 8; ++v) {
        int row = mBase + 16 * i + v + ((lane >> 4) << 3);
        int col = nBase + 16 * j + (lane & 15);
        float val = fmaxf(0.f, tanhf(acc[i][j][v]));
        out[(size_t)row * N_ROWS + col] = val;
      }
}

// ---------------- stage 3: per-row top-17 ----------------
__device__ __forceinline__ bool cand_gt(float av, int ai, float bv, int bi) {
  return (av > bv) || (av == bv && ai < bi);   // jax top_k tie-break: lower index wins
}

__global__ void __launch_bounds__(256) k_topk(const float* __restrict__ A,
                                              int* __restrict__ topIdx,
                                              float* __restrict__ topVal,
                                              float* __restrict__ rowSum) {
  __shared__ float shv[256 * TOPK];
  __shared__ int   shi[256 * TOPK];
  int row = blockIdx.x;
  int tid = threadIdx.x;
  const float* r = A + (size_t)row * N_ROWS;

  float lv[TOPK];
  int   li[TOPK];
#pragma unroll
  for (int t = 0; t < TOPK; ++t) { lv[t] = -1.0f; li[t] = 0x7fffffff; }

  for (int j = tid; j < N_ROWS; j += 256) {
    float v = r[j];
    if (cand_gt(v, j, lv[TOPK - 1], li[TOPK - 1])) {
      lv[TOPK - 1] = v; li[TOPK - 1] = j;
#pragma unroll
      for (int t = TOPK - 1; t > 0; --t) {
        if (cand_gt(lv[t], li[t], lv[t - 1], li[t - 1])) {
          float fv = lv[t]; lv[t] = lv[t - 1]; lv[t - 1] = fv;
          int   ii = li[t]; li[t] = li[t - 1]; li[t - 1] = ii;
        }
      }
    }
  }
#pragma unroll
  for (int t = 0; t < TOPK; ++t) { shv[tid * TOPK + t] = lv[t]; shi[tid * TOPK + t] = li[t]; }
  __syncthreads();

  for (int s = 128; s > 0; s >>= 1) {
    if (tid < s) {
      float av[TOPK], bv[TOPK], mv[TOPK];
      int   ai[TOPK], bi[TOPK], mi[TOPK];
#pragma unroll
      for (int t = 0; t < TOPK; ++t) {
        av[t] = shv[tid * TOPK + t];       ai[t] = shi[tid * TOPK + t];
        bv[t] = shv[(tid + s) * TOPK + t]; bi[t] = shi[(tid + s) * TOPK + t];
      }
      int pa = 0, pb = 0;
      for (int t = 0; t < TOPK; ++t) {
        bool takeA = cand_gt(av[pa], ai[pa], bv[pb], bi[pb]);
        if (takeA) { mv[t] = av[pa]; mi[t] = ai[pa]; ++pa; }
        else       { mv[t] = bv[pb]; mi[t] = bi[pb]; ++pb; }
      }
#pragma unroll
      for (int t = 0; t < TOPK; ++t) { shv[tid * TOPK + t] = mv[t]; shi[tid * TOPK + t] = mi[t]; }
    }
    __syncthreads();
  }

  if (tid < TOPK) {
    int   idx = shi[tid];
    float v   = shv[tid];
    if (idx == row) v = 0.f;              // diagonal removed from mask
    topIdx[row * TOPK + tid] = idx;
    topVal[row * TOPK + tid] = v;
  }
  if (tid == 0) {
    float s = 0.f;
#pragma unroll
    for (int t = 0; t < TOPK; ++t) s += (shi[t] == row) ? 0.f : shv[t];
    rowSum[row] = s;
  }
}

// ---------------- stage 4: deterministic global sum ----------------
__global__ void k_reduce(const float* __restrict__ rowSum, float* __restrict__ total) {
  __shared__ float sh[256];
  int tid = threadIdx.x;
  float s = 0.f;
  for (int i = tid; i < N_ROWS; i += 256) s += rowSum[i];
  sh[tid] = s;
  __syncthreads();
  for (int st = 128; st > 0; st >>= 1) {
    if (tid < st) sh[tid] += sh[tid + st];
    __syncthreads();
  }
  if (tid == 0) *total = sh[0];
}

// ---------------- stage 5: write masked + scaled output ----------------
__global__ void __launch_bounds__(256) k_finalize(const int* __restrict__ topIdx,
                                                  const float* __restrict__ topVal,
                                                  const float* __restrict__ total,
                                                  float* __restrict__ out) {
  __shared__ int   sidx[TOPK];
  __shared__ float sval[TOPK];
  int row = blockIdx.x;
  int tid = threadIdx.x;
  float mean = (*total) / (16.0f * (float)N_ROWS);
  float inv  = 1.0f / mean;
  if (tid < TOPK) {
    sidx[tid] = topIdx[row * TOPK + tid];
    sval[tid] = topVal[row * TOPK + tid] * inv;
  }
  __syncthreads();
  float* r = out + (size_t)row * N_ROWS;
  for (int j = tid; j < N_ROWS; j += 256) {
    float v = 0.f;
#pragma unroll
    for (int t = 0; t < TOPK; ++t)
      if (sidx[t] == j) v = sval[t];
    r[j] = v;
  }
}

// ---------------- launch ----------------
extern "C" void kernel_launch(void* const* d_in, const int* in_sizes, int n_in,
                              void* d_out, int out_size, void* d_ws, size_t ws_size,
                              hipStream_t stream) {
  const float* x   = (const float*)d_in[0];
  const float* lin = (const float*)d_in[1];
  float* out = (float*)d_out;

  char* ws = (char*)d_ws;
  size_t o = 0;
  bf16_t* xb   = (bf16_t*)(ws + o); o += (size_t)N_ROWS * C_DIM * sizeof(bf16_t);  // 4 MB
  bf16_t* linT = (bf16_t*)(ws + o); o += (size_t)C_DIM * C_DIM * sizeof(bf16_t);   // 128 KB
  bf16_t* hb   = (bf16_t*)(ws + o); o += (size_t)N_ROWS * C_DIM * sizeof(bf16_t);  // 4 MB
  int*   tIdx  = (int*)  (ws + o);  o += (size_t)N_ROWS * TOPK * sizeof(int);
  float* tVal  = (float*)(ws + o);  o += (size_t)N_ROWS * TOPK * sizeof(float);
  float* rSum  = (float*)(ws + o);  o += (size_t)N_ROWS * sizeof(float);
  float* total = (float*)(ws + o);  o += 256;

  k_convert_x   <<<(N_ROWS * C_DIM) / 256, 256, 0, stream>>>(x, xb, N_ROWS * C_DIM);
  k_convert_linT<<<(C_DIM * C_DIM) / 256, 256, 0, stream>>>(lin, linT);

  k_gemm_h<<<1024, 256, 0, stream>>>(xb, linT, hb);

  dim3 g2(N_ROWS / 128, N_ROWS / 128);   // 64 x 64 blocks, 128x128 per block
  k_gemm_sim<<<g2, 256, GEMM_LDS_BYTES, stream>>>(hb, out);

  k_topk    <<<N_ROWS, 256, 0, stream>>>(out, tIdx, tVal, rSum);
  k_reduce  <<<1, 256, 0, stream>>>(rSum, total);
  k_finalize<<<N_ROWS, 256, 0, stream>>>(tIdx, tVal, total, out);
}
